// APGANet_65094524338381
// MI455X (gfx1250) — matrix operand order repaired and checked by reading, not compiled, not run on hardware
//
#include <hip/hip_runtime.h>
#include <math.h>

typedef __attribute__((ext_vector_type(2)))  float  v2f;
typedef __attribute__((ext_vector_type(8)))  float  v8f;
typedef __attribute__((ext_vector_type(16))) __bf16 v16bf;
typedef __attribute__((ext_vector_type(8)))  __bf16 v8bf;

#define BSZ 2
#define CC 256
#define SS 32768                 // D*H*W = 32^3
#define NTOT (BSZ*SS)            // 65536 voxel columns
#define BCS ((size_t)BSZ*CC*SS)  // one (bs,256,S) tensor = 16,777,216 elems
#define SCALE_ 0.25f             // 16^-0.5
#define WTAB_ELEMS (6*65536)     // 3 matrices x {hi,lo} x 256*256 bf16

__device__ __forceinline__ float sigmoidf_(float x) { return 1.0f / (1.0f + __expf(-x)); }

#define WMMA_BF16(A, B, Cacc) \
    __builtin_amdgcn_wmma_f32_16x16x32_bf16(false, (A), false, (B), (short)0, (Cacc), false, false)

// A-fragment: 16 contiguous bf16 split as [K kc..kc+7][K kc+16..kc+23] (global)
__device__ __forceinline__ v16bf ldA(const __bf16* p) {
    v8bf a0 = *reinterpret_cast<const v8bf*>(p);
    v8bf a1 = *reinterpret_cast<const v8bf*>(p + 16);
    return __builtin_shufflevector(a0, a1, 0,1,2,3,4,5,6,7,8,9,10,11,12,13,14,15);
}
// B-fragment: 16 bf16 split as [K kb..kb+7][K kb+16..kb+23] at fixed voxel (LDS)
__device__ __forceinline__ v16bf ldB(const __bf16* p) {
    v8bf b0 = *reinterpret_cast<const v8bf*>(p);
    v8bf b1 = *reinterpret_cast<const v8bf*>(p + 16);
    return __builtin_shufflevector(b0, b1, 0,1,2,3,4,5,6,7,8,9,10,11,12,13,14,15);
}

// ---------------------------------------------------------------------------
// Kernel 0: split Wq/Wk/Wv into bf16 hi/lo tables (Markidis fp32-via-bf16x3)
// ---------------------------------------------------------------------------
__global__ __launch_bounds__(256)
void split_weights_kernel(const float* __restrict__ Wq, const float* __restrict__ Wk,
                          const float* __restrict__ Wv, __bf16* __restrict__ wtab)
{
    const int idx = blockIdx.x * 256 + threadIdx.x;     // < 3*65536
    const int mat = idx >> 16;
    const int e   = idx & 65535;
    const float* W = (mat == 0) ? Wq : (mat == 1) ? Wk : Wv;
    const float f  = W[e];
    const __bf16 h = (__bf16)f;
    const __bf16 l = (__bf16)(f - (float)h);
    wtab[(size_t)mat * 65536 + e]       = h;
    wtab[(size_t)(3 + mat) * 65536 + e] = l;
}

// ---------------------------------------------------------------------------
// Kernel 1: fused q/k/v GEMM (bf16x3 WMMA, fp32-accurate) + per-head softmax
// + SE convs + prior heads. One workgroup = 16 voxels x all 256 channels.
// ---------------------------------------------------------------------------
__global__ __launch_bounds__(256)
void fused_attn_se_kernel(
    const float* __restrict__ x, const __bf16* __restrict__ wtab,
    const float* __restrict__ bq, const float* __restrict__ bk, const float* __restrict__ bv,
    const float* __restrict__ Wq1p, const float* __restrict__ bq1p,
    const float* __restrict__ Wq2p, const float* __restrict__ bq2p,
    const float* __restrict__ se1_w1, const float* __restrict__ se1_b1,
    const float* __restrict__ se1_w2, const float* __restrict__ se1_b2,
    const float* __restrict__ se2_w1, const float* __restrict__ se2_b1,
    const float* __restrict__ se2_w2, const float* __restrict__ se2_b2,
    float* __restrict__ outs1, float* __restrict__ outs2,
    float* __restrict__ sup1,  float* __restrict__ sup2)
{
    __shared__ __align__(16) __bf16 Xhi[16][272];  // x tile hi, [voxel][K], padded
    __shared__ __align__(16) __bf16 Xlo[16][272];  // x tile lo
    __shared__ float Qs[256][16];                  // q (post bias) for sup heads
    __shared__ float Os[256][16];                  // attn*v output tile
    __shared__ float Ts[2][16][16];                // SE hidden (branch,16,voxel)

    const int tid  = threadIdx.x;
    const int lane = tid & 31;
    const int wv   = tid >> 5;

    const int col0 = blockIdx.x * 16;     // first voxel column
    const int b    = col0 >> 15;          // batch (S = 2^15)
    const int s0   = col0 & (SS - 1);     // spatial base

    // ---- stage X tile: K=256 rows x 16 voxels, f32 -> bf16 hi/lo in LDS
    {
        const int n     = tid & 15;
        const int kbase = (tid >> 4) * 16;
        const float* xp = x + (size_t)b * CC * SS + (size_t)s0 + n;
        for (int j = 0; j < 16; ++j) {
            const int k   = kbase + j;
            const float v = xp[(size_t)k * SS];
            const __bf16 h = (__bf16)v;
            Xhi[n][k] = h;
            Xlo[n][k] = (__bf16)(v - (float)h);
        }
    }
    __syncthreads();

    // ---- GEMM: wave wv computes channels [wv*32, wv*32+32) for q,k,v
    v8f accq0 = {}, accq1 = {}, acck0 = {}, acck1 = {}, accv0 = {}, accv1 = {};

    const int mlo = lane & 15;            // A row within tile / B voxel column
    const int kcA = (lane >> 4) << 3;     // A: K base 0 / 8  (+16 for 2nd half)
    const int kbB = (lane >> 4) << 3;     // B: K base 0 / 8  (+16 for 2nd half)
    const size_t rA0 = (size_t)(wv * 32 + mlo) * 256 + kcA;
    const size_t rA1 = rA0 + 16 * 256;

    const __bf16* wqh = wtab;
    const __bf16* wkh = wtab + 1 * 65536;
    const __bf16* wvh = wtab + 2 * 65536;
    const __bf16* wql = wtab + 3 * 65536;
    const __bf16* wkl = wtab + 4 * 65536;
    const __bf16* wvl = wtab + 5 * 65536;

    for (int kch = 0; kch < 8; ++kch) {
        const int ko = kch * 32;
        const v16bf bh = ldB(&Xhi[mlo][ko + kbB]);
        const v16bf bl = ldB(&Xlo[mlo][ko + kbB]);

        v16bf ah, al;
        // ---- q
        ah = ldA(wqh + rA0 + ko); al = ldA(wql + rA0 + ko);
        accq0 = WMMA_BF16(ah, bh, accq0);
        accq0 = WMMA_BF16(al, bh, accq0);
        accq0 = WMMA_BF16(ah, bl, accq0);
        ah = ldA(wqh + rA1 + ko); al = ldA(wql + rA1 + ko);
        accq1 = WMMA_BF16(ah, bh, accq1);
        accq1 = WMMA_BF16(al, bh, accq1);
        accq1 = WMMA_BF16(ah, bl, accq1);
        // ---- k
        ah = ldA(wkh + rA0 + ko); al = ldA(wkl + rA0 + ko);
        acck0 = WMMA_BF16(ah, bh, acck0);
        acck0 = WMMA_BF16(al, bh, acck0);
        acck0 = WMMA_BF16(ah, bl, acck0);
        ah = ldA(wkh + rA1 + ko); al = ldA(wkl + rA1 + ko);
        acck1 = WMMA_BF16(ah, bh, acck1);
        acck1 = WMMA_BF16(al, bh, acck1);
        acck1 = WMMA_BF16(ah, bl, acck1);
        // ---- v
        ah = ldA(wvh + rA0 + ko); al = ldA(wvl + rA0 + ko);
        accv0 = WMMA_BF16(ah, bh, accv0);
        accv0 = WMMA_BF16(al, bh, accv0);
        accv0 = WMMA_BF16(ah, bl, accv0);
        ah = ldA(wvh + rA1 + ko); al = ldA(wvl + rA1 + ko);
        accv1 = WMMA_BF16(ah, bh, accv1);
        accv1 = WMMA_BF16(al, bh, accv1);
        accv1 = WMMA_BF16(ah, bl, accv1);
    }

    // ---- epilogue: bias + per-head softmax(q*k*scale) * v
    // C/D layout: vgpr r, lane<16 -> M=r (N=lane); lane>=16 -> M=8+r (N=lane-16)
    {
        const int nn   = lane & 15;
        const int hoff = (lane < 16) ? 0 : 8;
        for (int mt = 0; mt < 2; ++mt) {
            const v8f aq = mt ? accq1 : accq0;
            const v8f ak = mt ? acck1 : acck0;
            const v8f av = mt ? accv1 : accv0;
            const int base_m = wv * 32 + mt * 16;  // == head index * 16
            float vv[8], tv[8];
            float tmax = -3.4e38f;
            for (int r = 0; r < 8; ++r) {
                const int ch = base_m + hoff + r;
                const float qf = aq[r] + bq[ch];
                const float kf = ak[r] + bk[ch];
                vv[r] = av[r] + bv[ch];
                Qs[ch][nn] = qf;
                tv[r] = qf * kf * SCALE_;
                tmax = fmaxf(tmax, tv[r]);
            }
            tmax = fmaxf(tmax, __shfl_xor(tmax, 16, 32));  // join lane-halves of head
            float ev[8], es = 0.f;
            for (int r = 0; r < 8; ++r) { ev[r] = __expf(tv[r] - tmax); es += ev[r]; }
            es += __shfl_xor(es, 16, 32);
            const float inv = 1.0f / es;
            for (int r = 0; r < 8; ++r) {
                const int ch = base_m + hoff + r;
                Os[ch][nn] = ev[r] * inv * vv[r];
            }
        }
    }
    __syncthreads();

    // ---- SE stage 1 (128 -> 16, relu) for both branches + sup heads
    {
        const int g = tid >> 4, vx = tid & 15;
        float a1 = se1_b1[g], a2 = se2_b1[g];
        for (int c = 0; c < 128; ++c) {
            a1 += se1_w1[g * 128 + c] * Os[c][vx];
            a2 += se2_w1[g * 128 + c] * Os[128 + c][vx];
        }
        Ts[0][g][vx] = fmaxf(a1, 0.f);
        Ts[1][g][vx] = fmaxf(a2, 0.f);

        if (tid < 96) {                       // sup_q1: 6 outputs x 16 voxels
            float a = bq1p[g];
            for (int c = 0; c < 128; ++c) a += Wq1p[g * 128 + c] * Qs[c][vx];
            sup1[(size_t)b * 6 * SS + (size_t)g * SS + s0 + vx] = a;
        } else if (tid < 112) {               // sup_q2: 1 output x 16 voxels
            float a = bq2p[0];
            for (int c = 0; c < 128; ++c) a += Wq2p[c] * Qs[128 + c][vx];
            sup2[(size_t)b * SS + s0 + vx] = a;
        }
    }
    __syncthreads();

    // ---- SE stage 2 (16 -> 256, sigmoid) -> out_s maps
    {
        const int cg = tid >> 4, vx = tid & 15;
        const size_t sbase = (size_t)b * CC * SS + (size_t)s0 + vx;
        for (int i = 0; i < 16; ++i) {
            const int ch = cg * 16 + i;
            float a1 = se1_b2[ch], a2 = se2_b2[ch];
            for (int j = 0; j < 16; ++j) {
                a1 += se1_w2[ch * 16 + j] * Ts[0][j][vx];
                a2 += se2_w2[ch * 16 + j] * Ts[1][j][vx];
            }
            outs1[sbase + (size_t)ch * SS] = sigmoidf_(a1);
            outs2[sbase + (size_t)ch * SS] = sigmoidf_(a2);
        }
    }
}

// ---------------------------------------------------------------------------
// Kernel 2: per-(branch,b,ch) max/mean/min pooling over S (no atomics)
// ---------------------------------------------------------------------------
__global__ __launch_bounds__(256)
void pool_kernel(const float* __restrict__ outs, float* __restrict__ stats)
{
    __shared__ float smx[256], smn[256], ssm[256];
    const int id  = blockIdx.x;               // branch*512 + b*256 + ch
    const int tid = threadIdx.x;
    const float* p = outs + (size_t)id * SS;
    float mx = -3.4e38f, mn = 3.4e38f, sm = 0.f;
    for (int i = tid; i < SS; i += 256) {
        const float v = p[i];
        mx = fmaxf(mx, v); mn = fminf(mn, v); sm += v;
    }
    smx[tid] = mx; smn[tid] = mn; ssm[tid] = sm;
    __syncthreads();
    for (int off = 128; off > 0; off >>= 1) {
        if (tid < off) {
            smx[tid] = fmaxf(smx[tid], smx[tid + off]);
            smn[tid] = fminf(smn[tid], smn[tid + off]);
            ssm[tid] += ssm[tid + off];
        }
        __syncthreads();
    }
    if (tid == 0) {
        stats[id * 3 + 0] = smx[0];
        stats[id * 3 + 1] = ssm[0] * (1.0f / SS);
        stats[id * 3 + 2] = smn[0];
    }
}

// ---------------------------------------------------------------------------
// Kernel 3: Conv1d(3->1) over pooling stats + FC squeeze-excite -> gates
// ---------------------------------------------------------------------------
__global__ __launch_bounds__(256)
void gate_kernel(const float* __restrict__ stats,
                 const float* __restrict__ sq1_w, const float* __restrict__ sq1_b,
                 const float* __restrict__ sq2_w, const float* __restrict__ sq2_b,
                 const float* __restrict__ sec1_w1, const float* __restrict__ sec1_b1,
                 const float* __restrict__ sec1_w2, const float* __restrict__ sec1_b2,
                 const float* __restrict__ sec2_w1, const float* __restrict__ sec2_b1,
                 const float* __restrict__ sec2_w2, const float* __restrict__ sec2_b2,
                 float* __restrict__ gates)
{
    __shared__ float sq[256];
    __shared__ float h[16];
    const int branch = blockIdx.x >> 1;
    const int bb     = blockIdx.x & 1;
    const int ch     = threadIdx.x;

    const float* qw = branch ? sq2_w  : sq1_w;
    const float  qb = branch ? sq2_b[0] : sq1_b[0];
    const float* w1 = branch ? sec2_w1 : sec1_w1;
    const float* b1 = branch ? sec2_b1 : sec1_b1;
    const float* w2 = branch ? sec2_w2 : sec1_w2;
    const float* b2 = branch ? sec2_b2 : sec1_b2;

    const int sidx = (branch * 512 + bb * 256 + ch) * 3;
    sq[ch] = stats[sidx] * qw[0] + stats[sidx + 1] * qw[1] + stats[sidx + 2] * qw[2] + qb;
    __syncthreads();
    if (ch < 16) {
        float a = b1[ch];
        for (int c = 0; c < 256; ++c) a += w1[ch * 256 + c] * sq[c];
        h[ch] = fmaxf(a, 0.f);
    }
    __syncthreads();
    float a = b2[ch];
    for (int j = 0; j < 16; ++j) a += w2[ch * 16 + j] * h[j];
    gates[branch * 512 + bb * 256 + ch] = sigmoidf_(a);
}

// ---------------------------------------------------------------------------
// Kernel 4: out = gate[branch,b,ch] * out_s   (both branches, contiguous)
// ---------------------------------------------------------------------------
__global__ __launch_bounds__(256)
void final_gate_kernel(const float* __restrict__ outs, const float* __restrict__ gates,
                       float* __restrict__ out)
{
    const size_t idx  = (size_t)blockIdx.x * 256 + threadIdx.x;  // < 2*BCS
    const size_t cidx = idx >> 15;                               // branch*512 + b*256 + ch
    out[idx] = gates[cidx] * outs[idx];
}

// ---------------------------------------------------------------------------
extern "C" void kernel_launch(void* const* d_in, const int* in_sizes, int n_in,
                              void* d_out, int out_size, void* d_ws, size_t ws_size,
                              hipStream_t stream)
{
    (void)in_sizes; (void)n_in; (void)out_size; (void)ws_size;
    const float* x     = (const float*)d_in[0];
    const float* Wq    = (const float*)d_in[1];  const float* bq    = (const float*)d_in[2];
    const float* Wk    = (const float*)d_in[3];  const float* bk    = (const float*)d_in[4];
    const float* Wv    = (const float*)d_in[5];  const float* bv    = (const float*)d_in[6];
    const float* Wq1p  = (const float*)d_in[7];  const float* bq1p  = (const float*)d_in[8];
    const float* Wq2p  = (const float*)d_in[9];  const float* bq2p  = (const float*)d_in[10];
    const float* sq1_w = (const float*)d_in[11]; const float* sq1_b = (const float*)d_in[12];
    const float* sq2_w = (const float*)d_in[13]; const float* sq2_b = (const float*)d_in[14];
    const float* se1_w1 = (const float*)d_in[15]; const float* se1_b1 = (const float*)d_in[16];
    const float* se1_w2 = (const float*)d_in[17]; const float* se1_b2 = (const float*)d_in[18];
    const float* se2_w1 = (const float*)d_in[19]; const float* se2_b1 = (const float*)d_in[20];
    const float* se2_w2 = (const float*)d_in[21]; const float* se2_b2 = (const float*)d_in[22];
    const float* sec1_w1 = (const float*)d_in[23]; const float* sec1_b1 = (const float*)d_in[24];
    const float* sec1_w2 = (const float*)d_in[25]; const float* sec1_b2 = (const float*)d_in[26];
    const float* sec2_w1 = (const float*)d_in[27]; const float* sec2_b1 = (const float*)d_in[28];
    const float* sec2_w2 = (const float*)d_in[29]; const float* sec2_b2 = (const float*)d_in[30];

    float* ws    = (float*)d_ws;
    float* outs1 = ws;                    // BCS floats (branch1 sigmoid map)
    float* outs2 = ws + BCS;              // BCS floats (branch2 sigmoid map)
    float* stats = ws + 2 * BCS;          // 2*2*256*3 = 3072 floats
    float* gates = stats + 3072;          // 2*2*256   = 1024 floats
    __bf16* wtab = (__bf16*)(gates + 1024);  // 6*65536 bf16 (16B-aligned offset)

    float* out  = (float*)d_out;
    float* sup1 = out + 2 * BCS;                 // (2,6,S)
    float* sup2 = sup1 + (size_t)BSZ * 6 * SS;   // (2,1,S)

    split_weights_kernel<<<768, 256, 0, stream>>>(Wq, Wk, Wv, wtab);

    fused_attn_se_kernel<<<NTOT / 16, 256, 0, stream>>>(
        x, wtab, bq, bk, bv, Wq1p, bq1p, Wq2p, bq2p,
        se1_w1, se1_b1, se1_w2, se1_b2, se2_w1, se2_b1, se2_w2, se2_b2,
        outs1, outs2, sup1, sup2);

    pool_kernel<<<1024, 256, 0, stream>>>(outs1, stats);

    gate_kernel<<<4, 256, 0, stream>>>(stats, sq1_w, sq1_b, sq2_w, sq2_b,
                                       sec1_w1, sec1_b1, sec1_w2, sec1_b2,
                                       sec2_w1, sec2_b1, sec2_w2, sec2_b2, gates);

    final_gate_kernel<<<(unsigned)(2 * BCS / 256), 256, 0, stream>>>(outs1, gates, out);
}
